// Decoder_1803886264480
// MI455X (gfx1250) — matrix operand order repaired and checked by reading, not compiled, not run on hardware
//
#include <hip/hip_runtime.h>
#include <hip/hip_bf16.h>

// ---------------------------------------------------------------------------
// Dimensions (fixed by the reference)
// ---------------------------------------------------------------------------
#define BB   64
#define LL   49
#define ENC  2048
#define EE   512
#define DD   512
#define AA   512
#define VV   10000
#define TT   60
#define TM   59          // T-1 decode steps
#define XH   3072        // [emb(512) | gated att_vec(2048) | h(512)]
#define XW   2560        // emb + att_vec part (input to wih)
#define G4   2048        // 4*D gates
#define DG   2560        // [dproj(512) | gate_pre(2048)]

typedef __attribute__((ext_vector_type(16))) _Float16 v16h;
typedef __attribute__((ext_vector_type(8)))  float    v8f;

#define DEV __device__ __forceinline__

DEV float sigm_(float x) { return 1.0f / (1.0f + __expf(-x)); }

#define WMMA_F16(a, b, c) \
  __builtin_amdgcn_wmma_f32_16x16x32_f16(false, (a), false, (b), (short)0, (c), false, false)

// ---------------------------------------------------------------------------
// WMMA fragment helpers (contiguous k order, wave32)
// A: 16x32 f16 tile of row-major [M x lda]
// lane<16 -> M=lane ; VGPR v: K=2v(+16*half)
// -> per lane this is bytes [32*half .. 32*half+31] of the row: 2x b128 loads
// ---------------------------------------------------------------------------
DEV v16h load_a_frag(const _Float16* __restrict__ A, int lda, int m0, int k0) {
  const int lane = threadIdx.x & 31;
  const int m    = lane & 15;
  const int hf   = lane >> 4;
  const _Float16* p = A + (size_t)(m0 + m) * lda + k0;
  v16h a;
#pragma unroll
  for (int v = 0; v < 8; ++v) {
    int kk = 2 * v + 16 * hf;
    a[2 * v]     = p[kk];
    a[2 * v + 1] = p[kk + 1];
  }
  return a;
}

// ---------------------------------------------------------------------------
// Fragment-packed B layout: weights are pre-swizzled in setup so that each
// 32x16 (KxN) tile is 512 contiguous halves, lane-record-major:
//   packed[(kt*Nt + nt)*512 + lane*16 + e] = W[kt*32 + e + 16*(lane>>4)][nt*16 + (lane&15)]
// which matches load_a_frag's contiguous k order -> one contiguous v16h load.
// ---------------------------------------------------------------------------
DEV size_t bpack_off(int k, int n, int Nt) {
  int kt = k >> 5, kin = k & 31;
  int nt = n >> 4, nin = n & 15;
  int lane = nin | ((kin >> 4) << 4);
  return (((size_t)(kt * Nt + nt)) << 9) + ((size_t)lane << 4) + (size_t)(kin & 15);
}

DEV v16h load_b_packed(const _Float16* __restrict__ P, int Nt, int kt, int nt) {
  const int lane = threadIdx.x & 31;
  const _Float16* p = P + (((size_t)(kt * Nt + nt)) << 9) + ((size_t)lane << 4);
  return *(const v16h*)p;   // 32B contiguous -> 2x global_load_b128
}

// ---------------------------------------------------------------------------
// Core K-loop, software-pipelined 2 deep: while WMMA(k) runs, fragments for
// k+1 and k+2 are in flight (>=4 outstanding b128 loads at every WMMA).
// ---------------------------------------------------------------------------
DEV v8f gemm_kloop(const _Float16* __restrict__ A, int lda, int m0,
                   const _Float16* __restrict__ P, int Nt, int nt, int KT) {
  v8f acc = {};
  v16h a0 = load_a_frag(A, lda, m0, 0);
  v16h b0 = load_b_packed(P, Nt, 0, nt);
  v16h a1 = a0, b1 = b0;
  if (KT > 1) {
    a1 = load_a_frag(A, lda, m0, 32);
    b1 = load_b_packed(P, Nt, 1, nt);
  }
  if (KT > 2)
    __builtin_prefetch(P + (((size_t)(2 * Nt + nt)) << 9) +
                           ((size_t)(threadIdx.x & 31) << 4), 0, 3);
  for (int kt = 0; kt < KT - 2; ++kt) {
    v16h a2 = load_a_frag(A, lda, m0, (kt + 2) << 5);
    v16h b2 = load_b_packed(P, Nt, kt + 2, nt);
    acc = WMMA_F16(a0, b0, acc);
    a0 = a1; b0 = b1; a1 = a2; b1 = b2;
  }
  if (KT > 1) {
    acc = WMMA_F16(a0, b0, acc);
    a0 = a1; b0 = b1;
  }
  acc = WMMA_F16(a0, b0, acc);
  return acc;
}

// ---------------------------------------------------------------------------
// Generic WMMA GEMM: C[M x ldc] = A[M x lda(f16)] * Wpacked[K x N] + bias
// 128 threads = 4 waves; wave w -> m-tile (blockIdx.y*4 + w); blockIdx.x -> n-tile
// ---------------------------------------------------------------------------
__global__ void wmma_gemm_bias(const _Float16* __restrict__ A, int lda,
                               const _Float16* __restrict__ P, int N,
                               const float* __restrict__ bias,
                               float* __restrict__ C, int ldc, int K) {
  const int wave = threadIdx.x >> 5;
  const int m0 = (blockIdx.y * 4 + wave) * 16;
  const int nt = blockIdx.x;
  v8f acc = gemm_kloop(A, lda, m0, P, N >> 4, nt, K >> 5);
  const int lane = threadIdx.x & 31;
  const int n  = (nt << 4) + (lane & 15);
  const int hf = lane >> 4;
  const float bv = bias[n];
#pragma unroll
  for (int r = 0; r < 8; ++r)
    C[(size_t)(m0 + r + (hf << 3)) * ldc + n] = acc[r] + bv;
}

// ---------------------------------------------------------------------------
// pred = h16 @ fc_w16(packed) + fc_b, masked write directly into d_out[b][t][v]
// grid.x = 625 n-tiles, 128 threads
// ---------------------------------------------------------------------------
__global__ void pred_gemm(const _Float16* __restrict__ H, // xh16+2560, stride XH
                          const _Float16* __restrict__ Pfc,
                          const float* __restrict__ fcb,
                          float* __restrict__ out,
                          const int* __restrict__ lens, int t) {
  const int wave = threadIdx.x >> 5;
  const int m0 = wave * 16;
  const int nt = blockIdx.x;
  v8f acc = gemm_kloop(H, XH, m0, Pfc, VV >> 4, nt, DD >> 5);
  const int lane = threadIdx.x & 31;
  const int n  = (nt << 4) + (lane & 15);
  const int hf = lane >> 4;
  const float bv = fcb[n];
#pragma unroll
  for (int r = 0; r < 8; ++r) {
    int b = m0 + r + (hf << 3);
    bool act = (lens[b] - 1) > t;
    out[(size_t)(b * TM + t) * VV + n] = act ? (acc[r] + bv) : 0.0f;
  }
}

// ---------------------------------------------------------------------------
// Attention step: scores -> softmax -> att_vec*gate -> x16; one block per b
// ---------------------------------------------------------------------------
__global__ void attention_step(const float* __restrict__ input,
                               const float* __restrict__ encproj,
                               const float* __restrict__ dg,
                               const float* __restrict__ attw,
                               const float* __restrict__ attb,
                               const _Float16* __restrict__ emb16,
                               _Float16* __restrict__ xh16,
                               float* __restrict__ probs_out,
                               const int* __restrict__ lens, int t) {
  const int b = blockIdx.x;
  const int tid = threadIdx.x;            // 256 threads, 8 waves
  const int wave = tid >> 5, lane = tid & 31;
  __shared__ float s_scores[LL];
  __shared__ float s_prob[LL];
  const float* dproj = dg + b * DG;

  for (int l = wave; l < LL; l += 8) {
    const float* ep = encproj + (size_t)(b * LL + l) * AA;
    float sum = 0.f;
    for (int e = lane; e < AA; e += 32) {
      float v = ep[e] + dproj[e];
      v = v > 0.f ? v : 0.f;
      sum += v * attw[e];
    }
#pragma unroll
    for (int off = 16; off > 0; off >>= 1) sum += __shfl_xor(sum, off, 32);
    if (lane == 0) s_scores[l] = sum + attb[0];
  }
  __syncthreads();
  if (tid == 0) {
    float mx = s_scores[0];
    for (int l = 1; l < LL; ++l) mx = fmaxf(mx, s_scores[l]);
    float ss = 0.f;
    for (int l = 0; l < LL; ++l) { float e = __expf(s_scores[l] - mx); s_prob[l] = e; ss += e; }
    float inv = 1.f / ss;
    for (int l = 0; l < LL; ++l) s_prob[l] *= inv;
  }
  __syncthreads();
  const bool active = (lens[b] - 1) > t;
  for (int l = tid; l < LL; l += 256)
    probs_out[(size_t)(b * TM + t) * LL + l] = active ? s_prob[l] : 0.f;
  // att_vec (f32) * gate -> f16 into x
  for (int e = tid; e < ENC; e += 256) {
    float av = 0.f;
    const float* ip = input + (size_t)(b * LL) * ENC + e;
#pragma unroll 7
    for (int l = 0; l < LL; ++l) av += s_prob[l] * ip[(size_t)l * ENC];
    float gate = sigm_(dg[b * DG + AA + e]);
    xh16[(size_t)b * XH + EE + e] = (_Float16)(gate * av);
  }
  for (int e = tid; e < EE; e += 256)
    xh16[(size_t)b * XH + e] = emb16[(size_t)(b * TM + t) * EE + e];
}

// ---------------------------------------------------------------------------
// LSTM pointwise cell, masked state update; h -> f16 slot in xh16
// ---------------------------------------------------------------------------
__global__ void lstm_cell(const float* __restrict__ gates,
                          float* __restrict__ cstate,
                          _Float16* __restrict__ xh16,
                          const int* __restrict__ lens, int t) {
  const int idx = blockIdx.x * blockDim.x + threadIdx.x;  // 64*512
  const int b = idx >> 9, j = idx & 511;
  const float* g = gates + (size_t)b * G4;
  float i_ = g[j], f_ = g[DD + j], g_ = g[2 * DD + j], o_ = g[3 * DD + j];
  float c  = cstate[idx];
  float cn = sigm_(f_) * c + sigm_(i_) * tanhf(g_);
  float hn = sigm_(o_) * tanhf(cn);
  if ((lens[b] - 1) > t) {
    cstate[idx] = cn;
    xh16[(size_t)b * XH + XW + j] = (_Float16)hn;
  }
}

// ---------------------------------------------------------------------------
// Setup kernels
// ---------------------------------------------------------------------------
__global__ void cvt_f32_f16(const float* __restrict__ s, _Float16* __restrict__ d, int n) {
  int i = blockIdx.x * blockDim.x + threadIdx.x;
  if (i < n) d[i] = (_Float16)s[i];
}

// f32 row-major [K x N] -> f16 fragment-packed
__global__ void cvt_swizzle(const float* __restrict__ src, _Float16* __restrict__ dst,
                            int K, int N) {
  int i = blockIdx.x * blockDim.x + threadIdx.x;
  if (i >= K * N) return;
  int k = i / N, n = i % N;
  dst[bpack_off(k, n, N >> 4)] = (_Float16)src[i];
}

__global__ void pack_wcat(const float* __restrict__ daw, const float* __restrict__ avw,
                          _Float16* __restrict__ w) {
  int i = blockIdx.x * blockDim.x + threadIdx.x;      // DD*DG
  if (i >= DD * DG) return;
  int k = i / DG, n = i % DG;
  float v = (n < AA) ? daw[k * AA + n] : avw[k * ENC + (n - AA)];
  w[bpack_off(k, n, DG >> 4)] = (_Float16)v;
}

__global__ void pack_wihwhh(const float* __restrict__ wih, const float* __restrict__ whh,
                            _Float16* __restrict__ w) {
  int i = blockIdx.x * blockDim.x + threadIdx.x;      // XH*G4
  if (i >= XH * G4) return;
  int k = i / G4, n = i % G4;
  float v = (k < XW) ? wih[(size_t)k * G4 + n] : whh[(size_t)(k - XW) * G4 + n];
  w[bpack_off(k, n, G4 >> 4)] = (_Float16)v;
}

__global__ void pack_biases(const float* __restrict__ dab, const float* __restrict__ avb,
                            const float* __restrict__ bih, const float* __restrict__ bhh,
                            float* __restrict__ bcat, float* __restrict__ bsum) {
  int i = blockIdx.x * blockDim.x + threadIdx.x;
  if (i < DG) bcat[i] = (i < AA) ? dab[i] : avb[i - AA];
  if (i < G4) bsum[i] = bih[i] + bhh[i];
}

__global__ void embed_gather(const int* __restrict__ captions,
                             const float* __restrict__ embw,
                             _Float16* __restrict__ emb16) {
  int i = blockIdx.x * blockDim.x + threadIdx.x;      // BB*TM*EE
  if (i >= BB * TM * EE) return;
  int e = i % EE, r = i / EE;
  int tt = r % TM, b = r / TM;
  int cap = captions[b * TT + tt];
  emb16[i] = (_Float16)embw[(size_t)cap * EE + e];
}

__global__ void mean_kernel(const float* __restrict__ input, float* __restrict__ mean) {
  int b = blockIdx.x;
  for (int e = threadIdx.x; e < ENC; e += 256) {
    float s = 0.f;
    const float* p = input + (size_t)(b * LL) * ENC + e;
    for (int l = 0; l < LL; ++l) s += p[(size_t)l * ENC];
    mean[b * ENC + e] = s * (1.0f / (float)LL);
  }
}

__global__ void init_hc(const float* __restrict__ mean,
                        const float* __restrict__ hw, const float* __restrict__ hb,
                        const float* __restrict__ cw, const float* __restrict__ cb,
                        float* __restrict__ cstate, _Float16* __restrict__ xh16) {
  int idx = blockIdx.x * blockDim.x + threadIdx.x;    // 64*512
  int b = idx >> 9, j = idx & 511;
  const float* m = mean + b * ENC;
  float h = hb[j], c = cb[j];
  for (int k = 0; k < ENC; ++k) {
    float mv = m[k];
    h += mv * hw[(size_t)k * DD + j];
    c += mv * cw[(size_t)k * DD + j];
  }
  cstate[idx] = c;
  xh16[(size_t)b * XH + XW + j] = (_Float16)h;
}

// ---------------------------------------------------------------------------
// Host launcher
// ---------------------------------------------------------------------------
static inline size_t alup(size_t x) { return (x + 255) & ~(size_t)255; }

extern "C" void kernel_launch(void* const* d_in, const int* in_sizes, int n_in,
                              void* d_out, int out_size, void* d_ws, size_t ws_size,
                              hipStream_t stream) {
  (void)in_sizes; (void)n_in; (void)out_size; (void)ws_size;
  const float* input   = (const float*)d_in[0];
  const int*   caps    = (const int*)  d_in[1];
  const int*   lens    = (const int*)  d_in[2];
  const float* embw    = (const float*)d_in[3];
  const float* eaw     = (const float*)d_in[4];
  const float* eab     = (const float*)d_in[5];
  const float* daw     = (const float*)d_in[6];
  const float* dab     = (const float*)d_in[7];
  const float* attw    = (const float*)d_in[8];
  const float* attb    = (const float*)d_in[9];
  const float* hw      = (const float*)d_in[10];
  const float* hb      = (const float*)d_in[11];
  const float* cw      = (const float*)d_in[12];
  const float* cb      = (const float*)d_in[13];
  const float* avw     = (const float*)d_in[14];
  const float* avb     = (const float*)d_in[15];
  const float* wih     = (const float*)d_in[16];
  const float* whh     = (const float*)d_in[17];
  const float* bih     = (const float*)d_in[18];
  const float* bhh     = (const float*)d_in[19];
  const float* fcw     = (const float*)d_in[20];
  const float* fcb     = (const float*)d_in[21];

  float* out_pred  = (float*)d_out;                        // [B][TM][V]
  float* out_probs = out_pred + (size_t)BB * TM * VV;      // [B][TM][L]

  // ---- workspace carve-up ----
  char* w = (char*)d_ws;
  size_t off = 0;
  auto take = [&](size_t bytes) { void* p = w + off; off += alup(bytes); return p; };
  _Float16* input16  = (_Float16*)take((size_t)BB * LL * ENC * 2);   // row-major (A side)
  _Float16* eaw16    = (_Float16*)take((size_t)ENC * AA * 2);        // packed
  _Float16* wcat16   = (_Float16*)take((size_t)DD * DG * 2);         // packed
  _Float16* wihwhh16 = (_Float16*)take((size_t)XH * G4 * 2);         // packed
  _Float16* fcw16    = (_Float16*)take((size_t)DD * VV * 2);         // packed
  _Float16* emb16    = (_Float16*)take((size_t)BB * TM * EE * 2);
  float*    encproj  = (float*)   take((size_t)BB * LL * AA * 4);
  _Float16* xh16     = (_Float16*)take((size_t)BB * XH * 2);
  float*    dgbuf    = (float*)   take((size_t)BB * DG * 4);
  float*    gates    = (float*)   take((size_t)BB * G4 * 4);
  float*    cstate   = (float*)   take((size_t)BB * DD * 4);
  float*    meanb    = (float*)   take((size_t)BB * ENC * 4);
  float*    bcat     = (float*)   take((size_t)DG * 4);
  float*    bsum     = (float*)   take((size_t)G4 * 4);

  // ---- one-time setup ----
  {
    int n = BB * LL * ENC;
    cvt_f32_f16<<<(n + 255) / 256, 256, 0, stream>>>(input, input16, n);
    n = ENC * AA;
    cvt_swizzle<<<(n + 255) / 256, 256, 0, stream>>>(eaw, eaw16, ENC, AA);
    n = DD * VV;
    cvt_swizzle<<<(n + 255) / 256, 256, 0, stream>>>(fcw, fcw16, DD, VV);
    n = DD * DG;
    pack_wcat<<<(n + 255) / 256, 256, 0, stream>>>(daw, avw, wcat16);
    n = XH * G4;
    pack_wihwhh<<<(n + 255) / 256, 256, 0, stream>>>(wih, whh, wihwhh16);
    pack_biases<<<(DG + 255) / 256, 256, 0, stream>>>(dab, avb, bih, bhh, bcat, bsum);
    n = BB * TM * EE;
    embed_gather<<<(n + 255) / 256, 256, 0, stream>>>(caps, embw, emb16);
    mean_kernel<<<BB, 256, 0, stream>>>(input, meanb);
    init_hc<<<(BB * DD) / 256, 256, 0, stream>>>(meanb, hw, hb, cw, cb, cstate, xh16);
    // enc_proj = input @ enc_att_w + b : M=3136, K=2048, N=512 (WMMA)
    dim3 g(AA / 16, (BB * LL) / 64);
    wmma_gemm_bias<<<g, 128, 0, stream>>>(input16, ENC, eaw16, AA, eab,
                                          encproj, AA, ENC);
  }

  // ---- recurrence: 59 steps ----
  for (int t = 0; t < TM; ++t) {
    // [dproj | gate_pre] = h16 @ [dec_att_w | actv_w] + [b|b] : 64x512x2560
    wmma_gemm_bias<<<dim3(DG / 16, 1), 128, 0, stream>>>(
        xh16 + XW, XH, wcat16, DG, bcat, dgbuf, DG, DD);
    // attention + softmax + gated att_vec + x assembly
    attention_step<<<BB, 256, 0, stream>>>(input, encproj, dgbuf, attw, attb,
                                           emb16, xh16, out_probs, lens, t);
    // gates = [x|h] @ [wih;whh] + (bih+bhh) : 64x3072x2048
    wmma_gemm_bias<<<dim3(G4 / 16, 1), 128, 0, stream>>>(
        xh16, XH, wihwhh16, G4, bsum, gates, G4, XH);
    // LSTM cell + masked state update
    lstm_cell<<<(BB * DD) / 256, 256, 0, stream>>>(gates, cstate, xh16, lens, t);
    // pred = h16 @ fc_w + fc_b (masked) : 64x512x10000
    pred_gemm<<<VV / 16, 128, 0, stream>>>(xh16 + XW, fcw16, fcb, out_pred, lens, t);
  }
}
